// LongFormerSelfAttentionBlock_37641093382376
// MI455X (gfx1250) — matrix operand order repaired and checked by reading, not compiled, not run on hardware
//
#include <hip/hip_runtime.h>
#include <hip/hip_bf16.h>
#include <cstdint>

typedef __attribute__((ext_vector_type(16))) _Float16 v16h;
typedef __attribute__((ext_vector_type(8)))  float    v8f;
typedef __attribute__((ext_vector_type(4)))  float    v4f;

union Frag16 { v16h h; v4f f[2]; };

#define BB   2
#define S0L  4095
#define SSL  4096
#define DDM  512
#define HHN  8
#define DHL  64
#define WWL  128
#define NCC  32
#define DFF  2048
#define MRR  (BB*SSL)      /* 8192 rows */
#define NEGBIG (-1e9f)

// ---------------- fragment helpers (layouts per CDNA5 ISA 7.12.2) ----------------
__device__ __forceinline__ v16h load_frag2(const _Float16* base, int off0, int off1) {
  Frag16 u;
  u.f[0] = *(const v4f*)(base + off0);
  u.f[1] = *(const v4f*)(base + off1);
  return u.h;
}
// A 16x32 f16: lane holds row (lane&15); elems 0..7 -> K=k0+hi*8.., 8..15 -> K=k0+16+hi*8..
__device__ __forceinline__ v16h load_a(const _Float16* row, int k0, int hi) {
  return load_frag2(row, k0 + hi*8, k0 + 16 + hi*8);
}
// B 32x16 f16 (from B^T rows): lane holds col (lane&15); elems 0..7 -> K=k0+hi*8.., 8..15 -> K=k0+16+hi*8..
__device__ __forceinline__ v16h load_b(const _Float16* rowT, int k0, int hi) {
  return load_frag2(rowT, k0 + hi*8, k0 + 16 + hi*8);
}
__device__ __forceinline__ v8f wmma16(v16h a, v16h b, v8f c) {
  return __builtin_amdgcn_wmma_f32_16x16x32_f16(false, a, false, b, (short)0, c, false, false);
}
__device__ __forceinline__ float half_max(float v) {
  #pragma unroll
  for (int m = 1; m < 16; m <<= 1) v = fmaxf(v, __shfl_xor(v, m));
  return v;
}
__device__ __forceinline__ float half_sum(float v) {
  #pragma unroll
  for (int m = 1; m < 16; m <<= 1) v += __shfl_xor(v, m);
  return v;
}

// ---------------- prep: pad x -> xp (f32 + f16) ----------------
__global__ __launch_bounds__(256) void prep_xp_kernel(const float* __restrict__ x,
                                                      float* __restrict__ xp32,
                                                      _Float16* __restrict__ xp16) {
  size_t i = (size_t)blockIdx.x * 256 + threadIdx.x;
  if (i >= (size_t)MRR * DDM) return;
  int c = (int)(i % DDM);
  size_t rs = i / DDM;
  int s = (int)(rs % SSL);
  int b = (int)(rs / SSL);
  float v = (s < S0L) ? x[((size_t)b * S0L + s) * DDM + c] : 0.f;
  xp32[i] = v;
  xp16[i] = (_Float16)v;
}

// ---------------- weight transpose W[K,N] f32 -> WT[N,K] f16 ----------------
__global__ __launch_bounds__(256) void transpose_w_kernel(const float* __restrict__ W,
                                                          _Float16* __restrict__ WT,
                                                          int K, int N) {
  size_t i = (size_t)blockIdx.x * 256 + threadIdx.x;
  if (i >= (size_t)K * N) return;
  int n = (int)(i % N);
  int k = (int)(i / N);
  WT[(size_t)n * K + k] = (_Float16)W[i];
}

// ---------------- v [B,S,H,Dh] f32 -> vT [B,H,Dh,S] f16 ----------------
__global__ __launch_bounds__(256) void transpose_v_kernel(const float* __restrict__ v32,
                                                          _Float16* __restrict__ vT16) {
  size_t i = (size_t)blockIdx.x * 256 + threadIdx.x;
  if (i >= (size_t)MRR * DDM) return;
  int d = (int)(i % DHL);
  size_t t = i / DHL;
  int h = (int)(t % HHN);
  size_t t2 = t / HHN;
  int s = (int)(t2 % SSL);
  int b = (int)(t2 / SSL);
  vT16[(((size_t)(b * HHN + h) * DHL + d)) * SSL + s] = (_Float16)v32[i];
}

// ---------------- generic WMMA GEMM: out = act((A16 @ BT16^T + bias)*scale) [+res] ----------------
__global__ __launch_bounds__(256) void gemm_f16_kernel(
    const _Float16* __restrict__ A, const _Float16* __restrict__ BT,
    const float* __restrict__ bias, int M, int N, int K,
    float scale, int act,
    float* __restrict__ out32, _Float16* __restrict__ out16,
    const float* __restrict__ res32) {
  const int tid = threadIdx.x;
  const int lane = tid & 31, wave = tid >> 5;
  const int hi = lane >> 4, ln = lane & 15;
  const int m_base = blockIdx.y * 128 + (wave & 3) * 32;
  const int n_base = blockIdx.x * 64 + (wave >> 2) * 32;

  const v8f vzero = {0.f,0.f,0.f,0.f,0.f,0.f,0.f,0.f};
  v8f acc[2][2];
  acc[0][0] = vzero; acc[0][1] = vzero; acc[1][0] = vzero; acc[1][1] = vzero;

  const _Float16* arow0 = A + (size_t)(m_base + ln) * K;
  const _Float16* arow1 = A + (size_t)(m_base + 16 + ln) * K;
  const _Float16* brow0 = BT + (size_t)(n_base + ln) * K;
  const _Float16* brow1 = BT + (size_t)(n_base + 16 + ln) * K;

  for (int k0 = 0; k0 < K; k0 += 32) {
    __builtin_prefetch(arow0 + k0 + 128, 0, 0);
    __builtin_prefetch(brow0 + k0 + 128, 0, 0);
    v16h a0 = load_a(arow0, k0, hi);
    v16h a1 = load_a(arow1, k0, hi);
    v16h b0 = load_b(brow0, k0, hi);
    v16h b1 = load_b(brow1, k0, hi);
    acc[0][0] = wmma16(a0, b0, acc[0][0]);
    acc[0][1] = wmma16(a0, b1, acc[0][1]);
    acc[1][0] = wmma16(a1, b0, acc[1][0]);
    acc[1][1] = wmma16(a1, b1, acc[1][1]);
  }

  #pragma unroll
  for (int jj = 0; jj < 2; jj++) {
    int col = n_base + jj * 16 + ln;
    float bv = bias ? bias[col] : 0.f;
    #pragma unroll
    for (int ii = 0; ii < 2; ii++) {
      #pragma unroll
      for (int r = 0; r < 8; r++) {
        int row = m_base + ii * 16 + hi * 8 + r;
        float v = (acc[ii][jj][r] + bv) * scale;
        if (act == 1) v = 0.5f * v * (1.f + erff(v * 0.70710678118654752f));
        size_t o = (size_t)row * N + col;
        if (res32) v += res32[o];
        if (out32) out32[o] = v;
        if (out16) out16[o] = (_Float16)v;
      }
    }
  }
}

// ---------------- qg matvec: qg[b,:] = (xp[b,last] @ Wqg + bqg) * scale ----------------
__global__ __launch_bounds__(256) void qg_kernel(const float* __restrict__ xp32,
                                                 const float* __restrict__ Wqg,
                                                 const float* __restrict__ bqg,
                                                 float* __restrict__ qg) {
  int t = blockIdx.x * 256 + threadIdx.x;
  if (t >= BB * DDM) return;
  int n = t % DDM, b = t / DDM;
  const float* xr = xp32 + ((size_t)(b * SSL + SSL - 1)) * DDM;
  float a = 0.f;
  for (int k = 0; k < DDM; k++) a += xr[k] * Wqg[(size_t)k * DDM + n];
  qg[t] = (a + bqg[n]) * 0.125f;
}

// ---------------- windowed attention (flash-style, WMMA scores + WMMA PV) ----------------
__global__ __launch_bounds__(256) void window_attn_kernel(
    const _Float16* __restrict__ q16, const _Float16* __restrict__ k16,
    const float* __restrict__ v32, const _Float16* __restrict__ vT16,
    const unsigned char* __restrict__ mask8,
    float* __restrict__ attn32) {
  __shared__ __align__(16) _Float16 pst[8][16][40];   // per-wave prob staging
  const int c = blockIdx.x, h = blockIdx.y, b = blockIdx.z;
  const int tid = threadIdx.x, lane = tid & 31, wave = tid >> 5;
  const int hi = lane >> 4, ln = lane & 15;
  const int qbase = wave * 16;
  const v8f vzero = {0.f,0.f,0.f,0.f,0.f,0.f,0.f,0.f};

  // q A-fragments (K = 0..31, 32..63 of Dh)
  const _Float16* qfrow = q16 + ((size_t)(b * SSL + c * WWL + qbase + ln)) * DDM + h * DHL;
  v16h qA0 = load_a(qfrow, 0, hi);
  v16h qA1 = load_a(qfrow, 32, hi);

  v8f pv[4]; pv[0] = vzero; pv[1] = vzero; pv[2] = vzero; pv[3] = vzero;
  float mi[8], li[8];
  #pragma unroll
  for (int r = 0; r < 8; r++) { mi[r] = -1e30f; li[r] = 0.f; }

  const int kwin0 = c * WWL - WWL;

  for (int kt = 0; kt < 12; kt++) {
    int kb = kwin0 + kt * 32;
    // ---- scores: two 16x16 tiles over 32 keys ----
    v8f sc0, sc1;
    {
      int kp0 = kb + ln;
      int kp1 = kb + 16 + ln;
      int kc0 = min(max(kp0, 0), SSL - 1);
      int kc1 = min(max(kp1, 0), SSL - 1);
      const _Float16* kr0 = k16 + ((size_t)(b * SSL + kc0)) * DDM + h * DHL;
      const _Float16* kr1 = k16 + ((size_t)(b * SSL + kc1)) * DDM + h * DHL;
      v16h kb0a = load_b(kr0, 0, hi),  kb0b = load_b(kr0, 32, hi);
      v16h kb1a = load_b(kr1, 0, hi),  kb1b = load_b(kr1, 32, hi);
      sc0 = vzero; sc0 = wmma16(qA0, kb0a, sc0); sc0 = wmma16(qA1, kb0b, sc0);
      sc1 = vzero; sc1 = wmma16(qA0, kb1a, sc1); sc1 = wmma16(qA1, kb1b, sc1);
    }
    // key validity (band + not padded + not global); lane's key columns:
    int key0 = kb + ln;
    int key1 = kb + 16 + ln;
    bool in0 = (key0 >= 0) && (key0 < S0L);
    bool in1 = (key1 >= 0) && (key1 < S0L);
    bool ok0 = in0 && (mask8[(size_t)b * S0L + (in0 ? key0 : 0)] == 0);
    bool ok1 = in1 && (mask8[(size_t)b * S0L + (in1 ? key1 : 0)] == 0);

    #pragma unroll
    for (int r = 0; r < 8; r++) {
      int qpos = c * WWL + qbase + hi * 8 + r;
      int d0 = key0 - qpos, d1 = key1 - qpos;
      bool v0 = ok0 && (d0 >= -WWL) && (d0 <= WWL);
      bool v1 = ok1 && (d1 >= -WWL) && (d1 <= WWL);
      float s0 = sc0[r], s1 = sc1[r];
      float m0 = v0 ? s0 : NEGBIG;
      float m1 = v1 ? s1 : NEGBIG;
      float tmax = half_max(fmaxf(m0, m1));
      float nm = fmaxf(mi[r], tmax);
      float scl = __expf(mi[r] - nm);
      float p0 = v0 ? __expf(s0 - nm) : 0.f;
      float p1 = v1 ? __expf(s1 - nm) : 0.f;
      float psum = half_sum(p0 + p1);
      li[r] = li[r] * scl + psum;
      mi[r] = nm;
      #pragma unroll
      for (int j = 0; j < 4; j++) pv[j][r] *= scl;
      pst[wave][hi * 8 + r][ln]      = (_Float16)p0;
      pst[wave][hi * 8 + r][16 + ln] = (_Float16)p1;
    }
    asm volatile("s_wait_dscnt 0" ::: "memory");  // own-wave LDS stores -> loads

    // ---- PV: probs A-frag from LDS, v^T B-frags, K = 32 keys ----
    const _Float16* prow = &pst[wave][ln][0];
    v16h pA = load_frag2(prow, hi * 16, hi * 16 + 8);
    int ks = kb + hi * 16;
    int ksc = min(max(ks, 0), SSL - 16);
    #pragma unroll
    for (int j = 0; j < 4; j++) {
      const _Float16* vrow = vT16 + ((size_t)(b * HHN + h) * DHL + (j * 16 + ln)) * SSL;
      v16h vB = load_frag2(vrow, ksc, ksc + 8);
      pv[j] = wmma16(pA, vB, pv[j]);
    }
  }

  // ---- fold in the global key (k[S-1], v[S-1]) ----
  const _Float16* klast = k16 + ((size_t)(b * SSL + SSL - 1)) * DDM + h * DHL;
  #pragma unroll
  for (int r = 0; r < 8; r++) {
    int q = qbase + hi * 8 + r;
    const _Float16* qrow = q16 + ((size_t)(b * SSL + c * WWL + q)) * DDM + h * DHL;
    float g = 0.f;
    for (int d = 0; d < DHL; d++) g += (float)qrow[d] * (float)klast[d];
    float nm = fmaxf(mi[r], g);
    float scl = __expf(mi[r] - nm);
    float pg = __expf(g - nm);
    li[r] = li[r] * scl + pg;
    mi[r] = nm;
    #pragma unroll
    for (int j = 0; j < 4; j++) {
      float vl = v32[((size_t)(b * SSL + SSL - 1)) * DDM + h * DHL + j * 16 + ln];
      pv[j][r] = pv[j][r] * scl + pg * vl;
    }
  }

  // ---- write normalized output ----
  #pragma unroll
  for (int j = 0; j < 4; j++) {
    #pragma unroll
    for (int r = 0; r < 8; r++) {
      int row = c * WWL + qbase + hi * 8 + r;
      attn32[((size_t)(b * SSL + row)) * DDM + h * DHL + j * 16 + ln] = pv[j][r] / li[r];
    }
  }
}

// ---------------- global token scores: gs[b,h,s] = qg . kg ----------------
__global__ __launch_bounds__(256) void gscore_kernel(const float* __restrict__ qg,
                                                     const float* __restrict__ kg32,
                                                     const unsigned char* __restrict__ mask8,
                                                     float* __restrict__ gs) {
  size_t t = (size_t)blockIdx.x * 256 + threadIdx.x;
  if (t >= (size_t)BB * HHN * SSL) return;
  int s = (int)(t % SSL);
  size_t t2 = t / SSL;
  int h = (int)(t2 % HHN);
  int b = (int)(t2 / HHN);
  bool pad = (s < S0L) && (mask8[(size_t)b * S0L + s] != 0);
  float a = 0.f;
  for (int d = 0; d < DHL; d++)
    a += qg[b * DDM + h * DHL + d] * kg32[((size_t)(b * SSL + s)) * DDM + h * DHL + d];
  gs[t] = pad ? NEGBIG : a;
}

// ---------------- global token softmax + output (writes attn row S-1) ----------------
__global__ __launch_bounds__(256) void og_kernel(const float* __restrict__ gs,
                                                 const float* __restrict__ vg32,
                                                 float* __restrict__ attn32) {
  __shared__ float ps[SSL];
  __shared__ float red[256];
  int bh = blockIdx.x;
  int h = bh % HHN, b = bh / HHN;
  const float* row = gs + (size_t)bh * SSL;
  int tid = threadIdx.x;

  float mx = -1e30f;
  for (int s = tid; s < SSL; s += 256) mx = fmaxf(mx, row[s]);
  red[tid] = mx; __syncthreads();
  for (int o = 128; o > 0; o >>= 1) { if (tid < o) red[tid] = fmaxf(red[tid], red[tid + o]); __syncthreads(); }
  mx = red[0]; __syncthreads();

  float sm = 0.f;
  for (int s = tid; s < SSL; s += 256) { float e = __expf(row[s] - mx); ps[s] = e; sm += e; }
  red[tid] = sm; __syncthreads();
  for (int o = 128; o > 0; o >>= 1) { if (tid < o) red[tid] += red[tid + o]; __syncthreads(); }
  float tot = red[0]; __syncthreads();

  if (tid < DHL) {
    float acc = 0.f;
    for (int s = 0; s < SSL; s++)
      acc += ps[s] * vg32[((size_t)(b * SSL + s)) * DDM + h * DHL + tid];
    attn32[((size_t)(b * SSL + SSL - 1)) * DDM + h * DHL + tid] = acc / tot;
  }
}

// ---------------- residual + LayerNorm -> hn (f32 + f16) ----------------
__global__ __launch_bounds__(256) void add_ln_kernel(const float* __restrict__ xp32,
                                                     const float* __restrict__ attn32,
                                                     const float* __restrict__ ln_g,
                                                     const float* __restrict__ ln_b,
                                                     float* __restrict__ hn32,
                                                     _Float16* __restrict__ hn16) {
  __shared__ float red[256];
  int row = blockIdx.x, tid = threadIdx.x;
  size_t base = (size_t)row * DDM;
  float h0 = xp32[base + tid] + attn32[base + tid];
  float h1 = xp32[base + tid + 256] + attn32[base + tid + 256];

  red[tid] = h0 + h1; __syncthreads();
  for (int o = 128; o > 0; o >>= 1) { if (tid < o) red[tid] += red[tid + o]; __syncthreads(); }
  float mu = red[0] * (1.f / DDM); __syncthreads();

  float d0 = h0 - mu, d1 = h1 - mu;
  red[tid] = d0 * d0 + d1 * d1; __syncthreads();
  for (int o = 128; o > 0; o >>= 1) { if (tid < o) red[tid] += red[tid + o]; __syncthreads(); }
  float var = red[0] * (1.f / DDM);
  float rstd = rsqrtf(var + 1e-5f);

  float o0 = d0 * rstd * ln_g[tid] + ln_b[tid];
  float o1 = d1 * rstd * ln_g[tid + 256] + ln_b[tid + 256];
  hn32[base + tid] = o0;       hn16[base + tid] = (_Float16)o0;
  hn32[base + tid + 256] = o1; hn16[base + tid + 256] = (_Float16)o1;
}

// ---------------- drop pad row, write d_out ----------------
__global__ __launch_bounds__(256) void final_copy_kernel(const float* __restrict__ y32,
                                                         float* __restrict__ out) {
  size_t i = (size_t)blockIdx.x * 256 + threadIdx.x;
  if (i >= (size_t)BB * S0L * DDM) return;
  int c = (int)(i % DDM);
  size_t t = i / DDM;
  int s = (int)(t % S0L);
  int b = (int)(t / S0L);
  out[i] = y32[((size_t)(b * SSL + s)) * DDM + c];
}

// =====================================================================
extern "C" void kernel_launch(void* const* d_in, const int* in_sizes, int n_in,
                              void* d_out, int out_size, void* d_ws, size_t ws_size,
                              hipStream_t stream) {
  (void)in_sizes; (void)n_in; (void)out_size; (void)ws_size;
  const float* x     = (const float*)d_in[0];
  const unsigned char* mask8 = (const unsigned char*)d_in[1];
  const float* Wq  = (const float*)d_in[2];  const float* bq  = (const float*)d_in[3];
  const float* Wk  = (const float*)d_in[4];  const float* bk  = (const float*)d_in[5];
  const float* Wv  = (const float*)d_in[6];  const float* bv  = (const float*)d_in[7];
  const float* Wqg = (const float*)d_in[8];  const float* bqg = (const float*)d_in[9];
  const float* Wkg = (const float*)d_in[10]; const float* bkg = (const float*)d_in[11];
  const float* Wvg = (const float*)d_in[12]; const float* bvg = (const float*)d_in[13];
  const float* lng = (const float*)d_in[14]; const float* lnb = (const float*)d_in[15];
  const float* W1  = (const float*)d_in[16]; const float* b1  = (const float*)d_in[17];
  const float* W2  = (const float*)d_in[18]; const float* b2  = (const float*)d_in[19];

  char* p = (char*)d_ws;
  auto alloc = [&](size_t bytes) -> void* {
    void* r = (void*)p;
    p += (bytes + 255) & ~(size_t)255;
    return r;
  };
  const size_t AE = (size_t)MRR * DDM;   // activation elements
  float*    xp32   = (float*)alloc(AE * 4);
  _Float16* xp16   = (_Float16*)alloc(AE * 2);
  _Float16* wqT    = (_Float16*)alloc((size_t)DDM * DDM * 2);
  _Float16* wkT    = (_Float16*)alloc((size_t)DDM * DDM * 2);
  _Float16* wvT    = (_Float16*)alloc((size_t)DDM * DDM * 2);
  _Float16* wkgT   = (_Float16*)alloc((size_t)DDM * DDM * 2);
  _Float16* wvgT   = (_Float16*)alloc((size_t)DDM * DDM * 2);
  _Float16* w1T    = (_Float16*)alloc((size_t)DDM * DFF * 2);
  _Float16* w2T    = (_Float16*)alloc((size_t)DFF * DDM * 2);
  _Float16* q16    = (_Float16*)alloc(AE * 2);
  _Float16* k16    = (_Float16*)alloc(AE * 2);
  float*    v32    = (float*)alloc(AE * 4);
  _Float16* vT16   = (_Float16*)alloc(AE * 2);
  float*    kg32   = (float*)alloc(AE * 4);
  float*    vg32   = (float*)alloc(AE * 4);
  float*    attn32 = (float*)alloc(AE * 4);
  float*    qg     = (float*)alloc((size_t)BB * DDM * 4);
  float*    gs     = (float*)alloc((size_t)BB * HHN * SSL * 4);
  float*    hn32   = (float*)alloc(AE * 4);
  _Float16* hn16   = (_Float16*)alloc(AE * 2);
  _Float16* ffn1   = (_Float16*)alloc((size_t)MRR * DFF * 2);
  float*    y32    = (float*)alloc(AE * 4);

  auto nb = [](size_t n) { return (unsigned)((n + 255) / 256); };

  // 1) pad + f16 copy of x
  prep_xp_kernel<<<nb(AE), 256, 0, stream>>>(x, xp32, xp16);

  // 2) weight transposes to f16 [N][K]
  transpose_w_kernel<<<nb((size_t)DDM*DDM), 256, 0, stream>>>(Wq,  wqT,  DDM, DDM);
  transpose_w_kernel<<<nb((size_t)DDM*DDM), 256, 0, stream>>>(Wk,  wkT,  DDM, DDM);
  transpose_w_kernel<<<nb((size_t)DDM*DDM), 256, 0, stream>>>(Wv,  wvT,  DDM, DDM);
  transpose_w_kernel<<<nb((size_t)DDM*DDM), 256, 0, stream>>>(Wkg, wkgT, DDM, DDM);
  transpose_w_kernel<<<nb((size_t)DDM*DDM), 256, 0, stream>>>(Wvg, wvgT, DDM, DDM);
  transpose_w_kernel<<<nb((size_t)DDM*DFF), 256, 0, stream>>>(W1,  w1T,  DDM, DFF);
  transpose_w_kernel<<<nb((size_t)DFF*DDM), 256, 0, stream>>>(W2,  w2T,  DFF, DDM);

  // 3) projections (WMMA GEMMs)
  dim3 gp(DDM / 64, MRR / 128);
  gemm_f16_kernel<<<gp, 256, 0, stream>>>(xp16, wqT,  bq,  MRR, DDM, DDM, 0.125f, 0, nullptr, q16, nullptr);
  gemm_f16_kernel<<<gp, 256, 0, stream>>>(xp16, wkT,  bk,  MRR, DDM, DDM, 1.f,    0, nullptr, k16, nullptr);
  gemm_f16_kernel<<<gp, 256, 0, stream>>>(xp16, wvT,  bv,  MRR, DDM, DDM, 1.f,    0, v32,  nullptr, nullptr);
  gemm_f16_kernel<<<gp, 256, 0, stream>>>(xp16, wkgT, bkg, MRR, DDM, DDM, 1.f,    0, kg32, nullptr, nullptr);
  gemm_f16_kernel<<<gp, 256, 0, stream>>>(xp16, wvgT, bvg, MRR, DDM, DDM, 1.f,    0, vg32, nullptr, nullptr);

  // 4) v^T for PV WMMA
  transpose_v_kernel<<<nb(AE), 256, 0, stream>>>(v32, vT16);

  // 5) windowed attention
  dim3 ga(NCC, HHN, BB);
  window_attn_kernel<<<ga, 256, 0, stream>>>(q16, k16, v32, vT16, mask8, attn32);

  // 6) global token path
  qg_kernel<<<nb((size_t)BB * DDM), 256, 0, stream>>>(xp32, Wqg, bqg, qg);
  gscore_kernel<<<nb((size_t)BB * HHN * SSL), 256, 0, stream>>>(qg, kg32, mask8, gs);
  og_kernel<<<BB * HHN, 256, 0, stream>>>(gs, vg32, attn32);

  // 7) residual + LayerNorm
  add_ln_kernel<<<MRR, 256, 0, stream>>>(xp32, attn32, lng, lnb, hn32, hn16);

  // 8) FFN (WMMA GEMMs; GELU fused; residual fused into second GEMM)
  dim3 g1(DFF / 64, MRR / 128);
  gemm_f16_kernel<<<g1, 256, 0, stream>>>(hn16, w1T, b1, MRR, DFF, DDM, 1.f, 1, nullptr, ffn1, nullptr);
  dim3 g2(DDM / 64, MRR / 128);
  gemm_f16_kernel<<<g2, 256, 0, stream>>>(ffn1, w2T, b2, MRR, DDM, DFF, 1.f, 0, y32, nullptr, hn32);

  // 9) drop pad row
  final_copy_kernel<<<nb((size_t)BB * S0L * DDM), 256, 0, stream>>>(y32, (float*)d_out);
}
